// MaskedBroadcastingSynthesizer_58076547776870
// MI455X (gfx1250) — matrix operand order repaired and checked by reading, not compiled, not run on hardware
//
#include <hip/hip_runtime.h>
#include <stdint.h>

// ---------------- CDNA5 WMMA types ----------------
typedef __bf16 v16bf __attribute__((ext_vector_type(16)));
typedef float  v8f   __attribute__((ext_vector_type(8)));
typedef float  v4f   __attribute__((ext_vector_type(4)));   // native vector for NT builtins

#define KDIM        256
#define TILE_M      64
#define ROW_PAD     264                 // 256 bf16 + 8 pad -> bank-conflict-free
#define ROW_BYTES   (ROW_PAD * 2)       // 528 B, 16B aligned
#define BS_BYTES    (256 * ROW_BYTES)   // staged B^T (bf16, padded rows): 135168
#define AS_BYTES    (TILE_M * ROW_BYTES)// one 64-row tile: 33792

__device__ __forceinline__ unsigned short f2bf(float f) {
    union { float f; uint32_t u; } x; x.f = f;
    uint32_t u = x.u;
    return (unsigned short)((u + 0x7FFFu + ((u >> 16) & 1u)) >> 16); // RNE
}

union FragBF { uint4 q[2]; v16bf v; };
union AccF   { v8f v; float f[8]; };

// A-matrix 16x32 bf16 fragment (ISA 7.12.2 layout):
// lane<16: M=lane, K0..7 @ +0..15, K16..23 @ +32..47
// lane>=16: M=lane-16, K8..15 @ +16..31, K24..31 @ +48..63
__device__ __forceinline__ v16bf load_a_frag(const char* base, int mBase, int k, int lane) {
    int row = mBase + (lane & 15);
    int off = row * ROW_BYTES + ((lane >> 4) << 4) + (k << 6);
    FragBF fr;
    fr.q[0] = *(const uint4*)(base + off);
    fr.q[1] = *(const uint4*)(base + off + 32);
    return fr.v;
}

// B-matrix 32x16 bf16 fragment, sourced from row-major B^T rows:
// lane<16: N=lane, K0..7 @ +0..15, K16..23 @ +32..47 ; lane>=16: N=lane-16, K8..15 @ +16..31, K24..31 @ +48..63
__device__ __forceinline__ v16bf load_b_frag(const char* base, int nBase, int k, int lane) {
    int n   = nBase + (lane & 15);
    int off = n * ROW_BYTES + ((lane >> 4) << 4) + (k << 6);
    FragBF fr;
    fr.q[0] = *(const uint4*)(base + off);
    fr.q[1] = *(const uint4*)(base + off + 32);
    return fr.v;
}

// Plain 64x256 @ 256x256 tile: 8 waves, each 16-row M strip x 128-col N half.
__device__ __forceinline__ void wmma_compute(const char* aBase, const char* bBase,
                                             int wave, int lane, AccF acc[8]) {
    const int mBase = (wave & 3) * 16;
    const int nHalf = (wave >> 2) * 128;
    #pragma unroll
    for (int nb = 0; nb < 8; ++nb)
        #pragma unroll
        for (int j = 0; j < 8; ++j) acc[nb].f[j] = 0.0f;

    #pragma unroll
    for (int k = 0; k < 8; ++k) {
        v16bf a = load_a_frag(aBase, mBase, k, lane);
        v16bf bf[8];
        #pragma unroll
        for (int nb = 0; nb < 8; ++nb)
            bf[nb] = load_b_frag(bBase, nHalf + nb * 16, k, lane);
        #pragma unroll
        for (int nb = 0; nb < 8; ++nb)
            acc[nb].v = __builtin_amdgcn_wmma_f32_16x16x32_bf16(
                false, a, false, bf[nb], (short)0, acc[nb].v, false, false);
    }
}

// Async global->LDS staging of bf16 [256][256] matrix into padded LDS rows.
// CDNA5 GLOBAL_LOAD_ASYNC_TO_LDS_B128 (ASYNCcnt); async_wait() before use.
__device__ __forceinline__ void stage_bt_async_issue(const unsigned short* __restrict__ BT,
                                                     char* Bs, int t) {
    #pragma unroll 4
    for (int i = 0; i < 32; ++i) {
        int e   = t + (i << 8);
        int row = e >> 5;                        // 32 uint4 per row
        int c   = e & 31;
        uint32_t lds = (uint32_t)(uintptr_t)(Bs + row * ROW_BYTES + (c << 4));
        uint64_t ga  = (uint64_t)(uintptr_t)BT + ((uint64_t)e << 4);
        asm volatile("global_load_async_to_lds_b128 %0, %1, off"
                     :: "v"(lds), "v"(ga) : "memory");
    }
}

__device__ __forceinline__ void async_wait() {
    asm volatile("s_wait_asynccnt 0x0" ::: "memory");
}

// One 1/16th chunk of an H tile: H[r][c..c+3] = relu(sW1[b,idx[r],c..] + pW1[f0+r,c..])
__device__ __forceinline__ void build_h_chunk(int i, int t, int b, int f0,
                                              const float* __restrict__ sW1,
                                              const float* __restrict__ pW1,
                                              const int* idxs, char* Hs) {
    int e4 = t + (i << 8);                       // float4 index in 64x256 tile
    int r  = e4 >> 6;
    int c4 = e4 & 63;
    const float4 sv = ((const float4*)(sW1 + (((size_t)b << 4) + idxs[r]) * KDIM))[c4];
    const float4 pv = ((const float4*)(pW1 + (size_t)(f0 + r) * KDIM))[c4];
    float h0 = sv.x + pv.x; h0 = h0 > 0.0f ? h0 : 0.0f;
    float h1 = sv.y + pv.y; h1 = h1 > 0.0f ? h1 : 0.0f;
    float h2 = sv.z + pv.z; h2 = h2 > 0.0f ? h2 : 0.0f;
    float h3 = sv.w + pv.w; h3 = h3 > 0.0f ? h3 : 0.0f;
    uint2 p;
    p.x = (uint32_t)f2bf(h0) | ((uint32_t)f2bf(h1) << 16);
    p.y = (uint32_t)f2bf(h2) | ((uint32_t)f2bf(h3) << 16);
    *(uint2*)(Hs + r * ROW_BYTES + (c4 << 3)) = p;
}

// Epilogue: C/D layout (VGPR j -> M=j lanes 0-15, M=j+8 lanes 16-31); NT stores.
__device__ __forceinline__ void store_tile(float* __restrict__ outB,
                                           const float* __restrict__ b2,
                                           int wave, int lane, AccF acc[8]) {
    const int nHalf = (wave >> 2) * 128;
    const int rbase = (wave & 3) * 16 + ((lane >> 4) << 3);
    for (int nb = 0; nb < 8; ++nb) {
        int col  = nHalf + nb * 16 + (lane & 15);
        float bv = b2[col];
        #pragma unroll
        for (int j = 0; j < 8; ++j)
            __builtin_nontemporal_store(acc[nb].f[j] + bv,
                                        &outB[(size_t)(rbase + j) * KDIM + col]);
    }
}

// -------- transpose + f32->bf16 convert: WT[n][k] = bf16(W[k][n]) --------
__global__ void k_transpose_bf16(const float* __restrict__ W, unsigned short* __restrict__ WT) {
    int n = blockIdx.x;
    int k = threadIdx.x;
    WT[n * 256 + k] = f2bf(W[k * 256 + n]);
}

// -------- generic C[M,256] = A[M,256] @ B  (B given as bf16 B^T) + bias --------
__global__ void __launch_bounds__(256, 1)
k_gemm_bf16(const float* __restrict__ A, const unsigned short* __restrict__ BT,
            const float* __restrict__ bias, float* __restrict__ C, int hasBias) {
    extern __shared__ char smem[];
    char* Bs = smem;
    char* As = smem + BS_BYTES;
    const int t  = threadIdx.x;
    const int m0 = blockIdx.x * TILE_M;

    stage_bt_async_issue(BT, Bs, t);     // overlaps with A-tile conversion below

    #pragma unroll 4
    for (int i = 0; i < 16; ++i) {
        int e4 = t + (i << 8);
        int r  = e4 >> 6;
        int c4 = e4 & 63;
        v4f av = __builtin_nontemporal_load(
            (const v4f*)(A + (size_t)(m0 + r) * KDIM) + c4);
        uint2 p;
        p.x = (uint32_t)f2bf(av.x) | ((uint32_t)f2bf(av.y) << 16);
        p.y = (uint32_t)f2bf(av.z) | ((uint32_t)f2bf(av.w) << 16);
        *(uint2*)(As + r * ROW_BYTES + (c4 << 3)) = p;
    }
    async_wait();
    __syncthreads();

    const int wave = t >> 5, lane = t & 31;
    AccF acc[8];
    wmma_compute(As, Bs, wave, lane, acc);

    const int nHalf = (wave >> 2) * 128;
    const int rbase = (wave & 3) * 16 + ((lane >> 4) << 3);
    for (int nb = 0; nb < 8; ++nb) {
        int col = nHalf + nb * 16 + (lane & 15);
        float bv = hasBias ? bias[col] : 0.0f;
        #pragma unroll
        for (int j = 0; j < 8; ++j)
            C[(size_t)(m0 + rbase + j) * KDIM + col] = acc[nb].f[j] + bv;  // re-read later: RT
    }
}

// -------- fused, 2 tiles per WG, software-pipelined --------
// While the WMMA loop runs over H0, the gather/relu/convert build of H1 is
// interleaved per k-step, hiding its L2 latency behind matrix issue.
// LDS = 132K (W2^T) + 2x33K (H0/H1) ~ 203 KB -> needs CDNA5's 320 KB WGP LDS.
__global__ void __launch_bounds__(256, 1)
k_fused_gather_gemm2(const float* __restrict__ mask,
                     const float* __restrict__ sW1,
                     const float* __restrict__ pW1,
                     const unsigned short* __restrict__ W2T,
                     const float* __restrict__ b2,
                     float* __restrict__ out) {
    extern __shared__ char smem[];
    char* Bs   = smem;                            // W2^T
    char* Hs0  = smem + BS_BYTES;                 // H tile 0
    char* Hs1  = Hs0 + AS_BYTES;                  // H tile 1
    int* idxs0 = (int*)(Hs1 + AS_BYTES);
    int* idxs1 = idxs0 + TILE_M;

    const int t  = threadIdx.x;
    const int b  = blockIdx.y;
    const int f0 = blockIdx.x * (2 * TILE_M);
    const int f1 = f0 + TILE_M;

    stage_bt_async_issue(W2T, Bs, t);    // overlaps with argmax + H0 build

    // argmax over 16 slots for both tiles (first max, matching jnp.argmax)
    if (t < 2 * TILE_M) {
        int tile = t >> 6, r = t & 63;
        const float* mp = mask + ((size_t)b << 4) * 4096 + f0 + tile * TILE_M + r;
        float best = __builtin_nontemporal_load(mp); int bi = 0;
        #pragma unroll
        for (int s = 1; s < 16; ++s) {
            float v = __builtin_nontemporal_load(mp + (size_t)s * 4096);
            if (v > best) { best = v; bi = s; }
        }
        (tile ? idxs1 : idxs0)[r] = bi;
    }
    __syncthreads();

    // build H0
    #pragma unroll 4
    for (int i = 0; i < 16; ++i)
        build_h_chunk(i, t, b, f0, sW1, pW1, idxs0, Hs0);
    async_wait();
    __syncthreads();

    const int wave = t >> 5, lane = t & 31;
    const int mBase = (wave & 3) * 16;
    const int nHalf = (wave >> 2) * 128;
    AccF acc[8];
    #pragma unroll
    for (int nb = 0; nb < 8; ++nb)
        #pragma unroll
        for (int j = 0; j < 8; ++j) acc[nb].f[j] = 0.0f;

    // ---- tile 0 WMMA, H1 build interleaved (2 chunks per k-step) ----
    #pragma unroll
    for (int k = 0; k < 8; ++k) {
        v16bf a = load_a_frag(Hs0, mBase, k, lane);
        v16bf bf[8];
        #pragma unroll
        for (int nb = 0; nb < 8; ++nb)
            bf[nb] = load_b_frag(Bs, nHalf + nb * 16, k, lane);
        #pragma unroll
        for (int nb = 0; nb < 8; ++nb)
            acc[nb].v = __builtin_amdgcn_wmma_f32_16x16x32_bf16(
                false, a, false, bf[nb], (short)0, acc[nb].v, false, false);
        build_h_chunk(2 * k,     t, b, f1, sW1, pW1, idxs1, Hs1);
        build_h_chunk(2 * k + 1, t, b, f1, sW1, pW1, idxs1, Hs1);
    }
    __syncthreads();                      // all H1 ds-stores visible

    float* outB = out + (((size_t)b * 4096) + f0) * KDIM;
    store_tile(outB, b2, wave, lane, acc);          // tile 0 (NT, fire-and-forget)

    // ---- tile 1 WMMA ----
    wmma_compute(Hs1, Bs, wave, lane, acc);
    store_tile(outB + (size_t)TILE_M * KDIM, b2, wave, lane, acc);   // tile 1
}

extern "C" void kernel_launch(void* const* d_in, const int* in_sizes, int n_in,
                              void* d_out, int out_size, void* d_ws, size_t ws_size,
                              hipStream_t stream) {
    const float* slots = (const float*)d_in[0];   // [64,16,256]
    const float* mask  = (const float*)d_in[1];   // [64,16,4096]
    const float* pos   = (const float*)d_in[2];   // [4096,256]
    const float* W1    = (const float*)d_in[3];   // [256,256]
    const float* b1    = (const float*)d_in[4];   // [256]
    const float* W2    = (const float*)d_in[5];   // [256,256]
    const float* b2    = (const float*)d_in[6];   // [256]
    float* out = (float*)d_out;                   // [64,4096,256]

    char* ws = (char*)d_ws;
    unsigned short* W1T = (unsigned short*)ws;                       // 128 KB
    unsigned short* W2T = (unsigned short*)(ws + 131072);            // 128 KB
    float* pW1 = (float*)(ws + 262144);                              // 4 MB  = pos@W1 + b1
    float* sW1 = (float*)(ws + 262144 + 4194304);                    // 1 MB  = slots@W1

    k_transpose_bf16<<<256, 256, 0, stream>>>(W1, W1T);
    k_transpose_bf16<<<256, 256, 0, stream>>>(W2, W2T);

    const size_t smemG = BS_BYTES + AS_BYTES;            // ~165 KB
    k_gemm_bf16<<<4096 / TILE_M, 256, smemG, stream>>>(pos,   W1T, b1, pW1, 1);
    k_gemm_bf16<<<1024 / TILE_M, 256, smemG, stream>>>(slots, W1T, b1, sW1, 0);

    const size_t smemF = BS_BYTES + 2 * AS_BYTES + 2 * TILE_M * sizeof(int);  // ~203 KB
    dim3 grid(4096 / (2 * TILE_M), 64);
    k_fused_gather_gemm2<<<grid, 256, smemF, stream>>>(mask, sW1, pW1, W2T, b2, out);
}
